// TriModalCrossAttention_11484742549628
// MI455X (gfx1250) — matrix operand order repaired and checked by reading, not compiled, not run on hardware
//
#include <hip/hip_runtime.h>

// ---------------------------------------------------------------------------
// TriModalCrossAttention for MI455X (gfx1250, wave32, WMMA).
//
// Roofline: ~168 GFLOP vs ~170 MB mandatory HBM traffic -> compute-bound.
// All matmuls use v_wmma_f32_16x16x32_f16 (f16 operands, f32 accumulate).
// Attention is flash-style (online softmax) so 4x2048x2048 scores are never
// materialized. LDS tile staging uses CDNA5 GLOBAL_LOAD_ASYNC_TO_LDS_B128
// (ASYNCcnt-tracked, no VGPR round-trip), finished with s_wait_asynccnt.
// ---------------------------------------------------------------------------

typedef _Float16 f16x8  __attribute__((ext_vector_type(8)));
typedef _Float16 f16x16 __attribute__((ext_vector_type(16)));
typedef float    f32x8  __attribute__((ext_vector_type(8)));

#define BB 4
#define SS 2048
#define DD 512
#define BSD (BB * SS * DD)   // 4,194,304 elements per o-tensor

#define WMMA_F16(a, b, c) \
  __builtin_amdgcn_wmma_f32_16x16x32_f16(false, (a), false, (b), (short)0, (c), false, false)

// ---- CDNA5 async memory->LDS copy (16B), tracked by ASYNCcnt --------------
static __device__ __forceinline__ void async_copy16(_Float16* lds, const _Float16* g) {
  unsigned la = (unsigned)(unsigned long long)(uintptr_t)lds;   // LDS offset = addr[31:0]
  unsigned long long ga = (unsigned long long)(uintptr_t)g;
  asm volatile("global_load_async_to_lds_b128 %0, %1, off" :: "v"(la), "v"(ga) : "memory");
}
static __device__ __forceinline__ void wait_async0() {
#if __has_builtin(__builtin_amdgcn_s_wait_asynccnt)
  __builtin_amdgcn_s_wait_asynccnt(0);
#else
  asm volatile("s_wait_asynccnt 0x0" ::: "memory");
#endif
}

// A fragment (16x32, M x K), source row-major with leading dim `ld` (halves).
// ISA layout: lanes 0-15 hold K 0..7 & 16..23 of row `lane`; lanes 16-31 hold
// K 8..15 & 24..31 of row `lane-16`.
static __device__ __forceinline__ f16x16 frag_a(const _Float16* base, int ld, int lane) {
  const _Float16* p = base + (lane & 15) * ld + ((lane >> 4) & 1) * 8;
  union { f16x16 v; f16x8 h[2]; } u;
  u.h[0] = *(const f16x8*)(p);
  u.h[1] = *(const f16x8*)(p + 16);
  return u.v;
}

// B fragment (32x16, K x N). B[k][n] supplied as row-major [n][k] buffer
// (weight rows / transposed V), leading dim `ld` halves. Lanes 0-15 hold
// K 0..7 & 16..23 of column `lane`; lanes 16-31 hold K 8..15 & 24..31 of column `lane-16`.
static __device__ __forceinline__ f16x16 frag_b(const _Float16* base, int ld, int lane) {
  const _Float16* p = base + (lane & 15) * ld + ((lane >> 4) & 1) * 8;
  union { f16x16 v; f16x8 h[2]; } u;
  u.h[0] = *(const f16x8*)(p);
  u.h[1] = *(const f16x8*)(p + 16);
  return u.v;
}

// --------------------------- conversion kernels ----------------------------

// x (f32 [8192][512]) -> xcat f16 [8192][1024], columns 0..511.
__global__ void cvt_x_kernel(const float* __restrict__ x, _Float16* __restrict__ xcat) {
  int i = blockIdx.x * blockDim.x + threadIdx.x;           // exact grid: 8192*512
  int row = i >> 9, col = i & 511;
  xcat[row * 1024 + col] = (_Float16)x[i];
}

__global__ void cvt_w_kernel(const float* __restrict__ w, _Float16* __restrict__ wh, int n) {
  int i = blockIdx.x * blockDim.x + threadIdx.x;
  if (i < n) wh[i] = (_Float16)w[i];
}

// ------------------------------- WMMA GEMM ---------------------------------
// Y[M=8192][N=512] = X[M][K] * W[N][K]^T + bias.  Block: 256 thr, tile 128x64.
// outh != null : write f16 (row stride 512)          [q/k/v projections]
// outf != null : write f32 to o_p region AND g region [output projections]
__global__ __launch_bounds__(256) void gemm_bias_kernel(
    const _Float16* __restrict__ X, int lda,
    const _Float16* __restrict__ W, const float* __restrict__ bias, int K,
    _Float16* __restrict__ outh, float* __restrict__ outf, int pidx) {
  __shared__ __attribute__((aligned(16))) _Float16 As[128 * 32];
  __shared__ __attribute__((aligned(16))) _Float16 Bs[64 * 32];
  const int tid = threadIdx.x, lane = tid & 31, w = tid >> 5;
  const int m0 = blockIdx.y * 128, n0 = blockIdx.x * 64;
  const int wrow = (w >> 1) * 32, wcol = (w & 1) * 32;
  f32x8 acc[2][2] = {};

  // per-thread staging coordinates (uniform trip counts, no divergence)
  const int ra0 = tid >> 2, ca0 = (tid & 3) * 8;          // A chunk 0 (rows 0..63)
  const int ra1 = ra0 + 64;                               // A chunk 1 (rows 64..127)

  for (int kc = 0; kc < K; kc += 32) {
    // async-stage A tile 128x32 and B tile 64x32 straight into LDS
    async_copy16(&As[ra0 * 32 + ca0], &X[(long)(m0 + ra0) * lda + kc + ca0]);
    async_copy16(&As[ra1 * 32 + ca0], &X[(long)(m0 + ra1) * lda + kc + ca0]);
    async_copy16(&Bs[ra0 * 32 + ca0], &W[(long)(n0 + ra0) * K + kc + ca0]);
    wait_async0();
    __syncthreads();

    f16x16 af[2], bf[2];
#pragma unroll
    for (int mt = 0; mt < 2; ++mt) af[mt] = frag_a(&As[(wrow + mt * 16) * 32], 32, lane);
#pragma unroll
    for (int nt = 0; nt < 2; ++nt) bf[nt] = frag_b(&Bs[(wcol + nt * 16) * 32], 32, lane);
#pragma unroll
    for (int mt = 0; mt < 2; ++mt)
#pragma unroll
      for (int nt = 0; nt < 2; ++nt)
        acc[mt][nt] = WMMA_F16(af[mt], bf[nt], acc[mt][nt]);
    __syncthreads();
  }

  const int clane = lane & 15, rbase = (lane >> 4) << 3;
#pragma unroll
  for (int mt = 0; mt < 2; ++mt)
#pragma unroll
    for (int nt = 0; nt < 2; ++nt) {
      int col = n0 + wcol + nt * 16 + clane;
      float bv = bias[col];
#pragma unroll
      for (int r = 0; r < 8; ++r) {
        int row = m0 + wrow + mt * 16 + rbase + r;
        float v = acc[mt][nt][r] + bv;
        if (outh) outh[(long)row * DD + col] = (_Float16)v;
        if (outf) {
          outf[(long)pidx * BSD + (long)row * DD + col] = v;    // o_p
          int b = row >> 11;                                    // row = b*2048+s
          long grow = (long)row + (long)b * 4096 + (long)pidx * 2048;
          outf[3L * BSD + grow * DD + col] = v;                 // g
        }
      }
    }
}

// ---------------------------- flash attention ------------------------------
// Block: 128 thr (4 waves), 16 queries x full D=512. Key tiles of 64.
// Each wave owns a 128-wide slice of D for the P*V accumulation.
__global__ __launch_bounds__(128) void flash_kernel(
    const _Float16* __restrict__ Q, const _Float16* __restrict__ Kh,
    const _Float16* __restrict__ Vh, _Float16* __restrict__ Oc, float scale) {
  __shared__ __attribute__((aligned(16))) _Float16 Qs[16 * 512];   // 16 KB
  __shared__ __attribute__((aligned(16))) _Float16 KVs[512 * 32];  // 32 KB (K chunk / V^T overlay)
  __shared__ __attribute__((aligned(16))) float    Ss[16 * 64];    // 4 KB
  __shared__ __attribute__((aligned(16))) _Float16 Avs[16 * 64];   // 2 KB
  __shared__ float m_s[16], l_s[16], sc_s[16];

  const int tid = threadIdx.x, lane = tid & 31, w = tid >> 5;
  const int qrow0 = blockIdx.y * SS + blockIdx.x * 16;

  // async-stage Q (16x512): 8 x 16B per thread, uniform
#pragma unroll
  for (int it = 0; it < 8; ++it) {
    int i = tid + it * 128;
    int r = i >> 6, c8 = (i & 63) * 8;
    async_copy16(&Qs[r * 512 + c8], &Q[(long)(qrow0 + r) * DD + c8]);
  }
  if (tid < 16) { m_s[tid] = -3.0e38f; l_s[tid] = 0.0f; }
  wait_async0();
  __syncthreads();

  f32x8 acc[8] = {};                                      // 16 x 128 per wave

  const int rk = tid >> 2, ck = (tid & 3) * 8;            // K-chunk staging coords
  for (int kt = 0; kt < SS / 64; ++kt) {
    const long key0 = (long)blockIdx.y * SS + (long)kt * 64;

    // ---- scores: S(16x64) = Q(16x512) * K_tile(64x512)^T, wave w -> 16 keys
    f32x8 sacc = {};
    for (int dc = 0; dc < DD; dc += 32) {
      async_copy16(&KVs[rk * 32 + ck], &Kh[(key0 + rk) * DD + dc + ck]);
      async_copy16(&KVs[(rk + 32) * 32 + ck], &Kh[(key0 + rk + 32) * DD + dc + ck]);
      wait_async0();
      __syncthreads();
      f16x16 af = frag_a(&Qs[dc], 512, lane);
      f16x16 bf = frag_b(&KVs[w * 16 * 32], 32, lane);
      sacc = WMMA_F16(af, bf, sacc);
      __syncthreads();
    }
    {
      const int clane = lane & 15, rbase = (lane >> 4) << 3;
#pragma unroll
      for (int r = 0; r < 8; ++r)
        Ss[(rbase + r) * 64 + w * 16 + clane] = sacc[r] * scale;
    }
    __syncthreads();

    // ---- online softmax (wave 0, one lane per query row)
    if (tid < 16) {
      const int r = tid;
      float rowmax = -3.0e38f;
      for (int j = 0; j < 64; ++j) rowmax = fmaxf(rowmax, Ss[r * 64 + j]);
      float mnew = fmaxf(m_s[r], rowmax);
      float sum = 0.0f;
      for (int j = 0; j < 64; ++j) {
        float p = __expf(Ss[r * 64 + j] - mnew);
        Avs[r * 64 + j] = (_Float16)p;
        sum += p;
      }
      float sc = __expf(m_s[r] - mnew);
      l_s[r] = l_s[r] * sc + sum;
      m_s[r] = mnew;
      sc_s[r] = sc;
    }
    __syncthreads();

    // ---- rescale running accumulators
    {
      const int rbase = (lane >> 4) << 3;
      float scv[8];
#pragma unroll
      for (int r = 0; r < 8; ++r) scv[r] = sc_s[rbase + r];
#pragma unroll
      for (int nt = 0; nt < 8; ++nt)
#pragma unroll
        for (int r = 0; r < 8; ++r) acc[nt][r] *= scv[r];
    }

    // ---- acc += P(16x64) * V_tile(64x512), in two 32-key passes
    for (int khf = 0; khf < 2; ++khf) {
      __syncthreads();
      for (int i = tid; i < 2048; i += 128) {             // V^T stage: [d][key]
        int key = i >> 6, c8 = (i & 63) * 8;
        f16x8 vv = *(const f16x8*)&Vh[(key0 + khf * 32 + key) * DD + c8];
#pragma unroll
        for (int e = 0; e < 8; ++e) KVs[(c8 + e) * 32 + key] = vv[e];
      }
      __syncthreads();
      f16x16 af = frag_a(&Avs[khf * 32], 64, lane);
#pragma unroll
      for (int nt = 0; nt < 8; ++nt) {
        f16x16 bf = frag_b(&KVs[(w * 128 + nt * 16) * 32], 32, lane);
        acc[nt] = WMMA_F16(af, bf, acc[nt]);
      }
    }
    __syncthreads();
  }

  // ---- finalize: divide by l, write c as f16 into xcat columns 512..1023
  {
    const int clane = lane & 15, rbase = (lane >> 4) << 3;
    float li[8];
#pragma unroll
    for (int r = 0; r < 8; ++r) li[r] = 1.0f / l_s[rbase + r];
#pragma unroll
    for (int nt = 0; nt < 8; ++nt)
#pragma unroll
      for (int r = 0; r < 8; ++r) {
        long row = qrow0 + rbase + r;
        int col = 512 + w * 128 + nt * 16 + clane;
        Oc[row * 1024 + col] = (_Float16)(acc[nt][r] * li[r]);
      }
  }
}

// ------------------------------- launcher ----------------------------------

extern "C" void kernel_launch(void* const* d_in, const int* in_sizes, int n_in,
                              void* d_out, int out_size, void* d_ws, size_t ws_size,
                              hipStream_t stream) {
  (void)in_sizes; (void)n_in; (void)out_size; (void)ws_size;

  // workspace layout (f16): xcat[3] | q/k/v[9] | Wqkv_h[9] | Wo_h[3]  (~128 MB)
  char* ws = (char*)d_ws;
  const size_t XCAT_B = (size_t)8192 * 1024 * 2;
  const size_t QKV_B  = (size_t)8192 * 512 * 2;
  const size_t WH_B   = (size_t)512 * 512 * 2;
  const size_t WOH_B  = (size_t)512 * 1024 * 2;

  _Float16* xcat[3]; _Float16* qkv[9]; _Float16* wh[9]; _Float16* woh[3];
  for (int p = 0; p < 3; ++p) xcat[p] = (_Float16*)(ws + p * XCAT_B);
  char* base = ws + 3 * XCAT_B;
  for (int i = 0; i < 9; ++i) qkv[i] = (_Float16*)(base + i * QKV_B);
  base += 9 * QKV_B;
  for (int i = 0; i < 9; ++i) wh[i] = (_Float16*)(base + i * WH_B);
  base += 9 * WH_B;
  for (int p = 0; p < 3; ++p) woh[p] = (_Float16*)(base + p * WOH_B);

  // f32 -> f16 conversions
  for (int m = 0; m < 3; ++m)
    cvt_x_kernel<<<dim3(16384), dim3(256), 0, stream>>>((const float*)d_in[m], xcat[m]);
  for (int i = 0; i < 9; ++i)   // d_in: 3=Wq1,5=Wk1,7=Wv1,9=Wq2,...
    cvt_w_kernel<<<dim3(1024), dim3(256), 0, stream>>>((const float*)d_in[3 + 2 * i], wh[i], 512 * 512);
  for (int p = 0; p < 3; ++p)   // d_in: 21=Wo1,23=Wo2,25=Wo3
    cvt_w_kernel<<<dim3(2048), dim3(256), 0, stream>>>((const float*)d_in[21 + 2 * p], woh[p], 512 * 1024);

  // QKV projections (read xcat cols 0..511, lda=1024)
  dim3 ggrid(512 / 64, 8192 / 128);
  for (int m = 0; m < 3; ++m)
    for (int j = 0; j < 3; ++j)  // j: 0=q, 1=k, 2=v; biases at 4+6m+2j
      gemm_bias_kernel<<<ggrid, 256, 0, stream>>>(
          xcat[m], 1024, wh[3 * m + j], (const float*)d_in[4 + 6 * m + 2 * j], 512,
          qkv[3 * m + j], nullptr, 0);

  // cross attention: c1=attn(q1,k2,v2), c2=attn(q2,k3,v3), c3=attn(q3,k1,v1)
  dim3 agrid(SS / 16, BB);
  const float scl = 0.044194173824159216f;  // 1/sqrt(512)
  for (int p = 0; p < 3; ++p) {
    int mm = (p + 1) % 3;
    flash_kernel<<<agrid, 128, 0, stream>>>(qkv[3 * p], qkv[3 * mm + 1], qkv[3 * mm + 2],
                                            xcat[p], scl);
  }

  // output projections: o_p = [x_p | c_p] * Wo_p^T + bo_p, dual-stored (o + g)
  for (int p = 0; p < 3; ++p)
    gemm_bias_kernel<<<ggrid, 256, 0, stream>>>(
        xcat[p], 1024, woh[p], (const float*)d_in[22 + 2 * p], 1024,
        nullptr, (float*)d_out, p);
}
